// GatedDeltaNet_80255758893685
// MI455X (gfx1250) — matrix operand order repaired and checked by reading, not compiled, not run on hardware
//
#include <hip/hip_runtime.h>
#include <hip/hip_bf16.h>
#include <math.h>

#define B_      2
#define T_      2048
#define HID_    2048
#define NH_     6
#define DK_     256
#define DV_     512
#define KEYD    (NH_*DK_)    // 1536
#define VALD    (NH_*DV_)    // 3072
#define BT_     (B_*T_)      // 4096
#define CONV_   4
#define EPS_    1e-5f
#define SCALE_  0.0625f      // DK^-0.5

typedef __attribute__((ext_vector_type(16))) __bf16 v16bf;
typedef __attribute__((ext_vector_type(8)))  __bf16 v8bf;
typedef __attribute__((ext_vector_type(8)))  float  v8f;

// ---------- small device helpers ----------
__device__ __forceinline__ float dev_silu(float x)    { return x / (1.0f + expf(-x)); }
__device__ __forceinline__ float dev_sigmoid(float x) { return 1.0f / (1.0f + expf(-x)); }
__device__ __forceinline__ float dev_softplus(float x){ return (x > 20.0f) ? x : log1pf(expf(x)); }

__device__ __forceinline__ unsigned int f2bf_bits(float f) {
  union { float f; unsigned int u; } a; a.f = f;
  return (a.u + 0x7FFFu + ((a.u >> 16) & 1u)) >> 16;
}
// pack two f32 -> two bf16 in one dword (lo = first element)
__device__ __forceinline__ unsigned int pack2bf(float lo, float hi) {
  return (f2bf_bits(hi) << 16) | (f2bf_bits(lo) & 0xFFFFu);
}

// xor-add across lanes via ds_bpermute (wave32)
__device__ __forceinline__ float bperm_xor_add(float x, int lane, int mask) {
  int idx = ((lane ^ mask) << 2);
  float y = __int_as_float(__builtin_amdgcn_ds_bpermute(idx, __float_as_int(x)));
  return x + y;
}
__device__ __forceinline__ float xor_reduce16(float x, int lane) {
  x = bperm_xor_add(x, lane, 1);
  x = bperm_xor_add(x, lane, 2);
  x = bperm_xor_add(x, lane, 4);
  x = bperm_xor_add(x, lane, 8);
  return x;
}
__device__ __forceinline__ float xor_reduce32(float x, int lane) {
  x = xor_reduce16(x, lane);
  x = bperm_xor_add(x, lane, 16);
  return x;
}

// =====================================================================
// 1) bf16 WMMA GEMM: C[M,N] = A[M,K] * B[K,N]   (fp32 in/out, bf16 MACs)
//    Block tile 64x128, 128 threads (4 waves); wave tile 16x128.
//    Register-software-pipelined fill + double-buffered LDS:
//      issue global loads(k+1) -> frag ds_loads(k) -> one dscnt wait ->
//      8 back-to-back WMMAs(k) -> wait loads, pack, ds_store(k+1) -> bar.
//    Phases fenced with sched_barrier so HBM latency hides under WMMA.
// =====================================================================
#define GTM 64
#define GTN 128
#define GTK 32
#define APAD 40   // A row stride (bf16): 80B, 16B-aligned
#define BPAD 56   // Bt row stride (bf16): 112B, 16B-aligned, 2-way conflict

__global__ __launch_bounds__(128)
void gdn_gemm_bf16_wmma(const float* __restrict__ A, const float* __restrict__ Bm,
                        float* __restrict__ C, int M, int N, int K) {
  __shared__ __attribute__((aligned(16))) __bf16 As[2][GTM][APAD];
  __shared__ __attribute__((aligned(16))) __bf16 Bt[2][GTN][BPAD];   // [n][k]
  const int tid  = threadIdx.x;
  const int lane = tid & 31;
  const int wave = tid >> 5;            // 0..3 -> 16-row strip
  const int half = lane >> 4;           // 0/1
  const int l15  = lane & 15;
  const int m0 = blockIdx.y * GTM;
  const int n0 = blockIdx.x * GTN;

  v8f acc[8] = {};

  float2 areg[8];
  float  breg_lo[16], breg_hi[16];

  // phase 1: issue global loads for tile at k0 into registers (no LDS yet)
  auto gload = [&](int k0) {
#pragma unroll
    for (int e = 0; e < 8; ++e) {
      int idx = tid + e * 128;          // pair index
      int r = idx >> 4, c2 = idx & 15;  // row, pair-col
      areg[e] = *(const float2*)&A[(size_t)(m0 + r) * K + (k0 + 2 * c2)];
    }
#pragma unroll
    for (int e = 0; e < 16; ++e) {
      int idx = tid + e * 128;            // pair index
      int n = idx & 127, kp = idx >> 7;   // col, pair-row (k = 2*kp)
      const size_t g = (size_t)(k0 + 2 * kp) * N + (n0 + n);
      breg_lo[e] = Bm[g];
      breg_hi[e] = Bm[g + N];
    }
  };
  // phase 2: pack bf16 pairs and commit to LDS buffer
  auto lstore = [&](int buf) {
#pragma unroll
    for (int e = 0; e < 8; ++e) {
      int idx = tid + e * 128;
      int r = idx >> 4, c2 = idx & 15;
      ((unsigned int*)&As[buf][r][0])[c2] = pack2bf(areg[e].x, areg[e].y);
    }
#pragma unroll
    for (int e = 0; e < 16; ++e) {
      int idx = tid + e * 128;
      int n = idx & 127, kp = idx >> 7;
      ((unsigned int*)&Bt[buf][n][0])[kp] = pack2bf(breg_lo[e], breg_hi[e]);
    }
  };

  const int nk = K / GTK;
  gload(0);
  lstore(0);
  __syncthreads();

  for (int kb = 0; kb < nk; ++kb) {
    const int cur = kb & 1;
    const bool hasnext = (kb + 1 < nk);
    if (hasnext) {
      gload((kb + 1) * GTK);             // issue; waited only after WMMAs
      if (kb + 2 < nk) {
        __builtin_prefetch(&A[(size_t)m0 * K + ((kb + 2) * GTK)], 0, 1);
        __builtin_prefetch(&Bm[(size_t)((kb + 2) * GTK) * N + n0], 0, 1);
      }
    }
    __builtin_amdgcn_sched_barrier(0);

    // ---- A fragment (ISA 7.12.2 16-bit A layout): two contiguous 8-elem runs
    const v8bf* arow = (const v8bf*)&As[cur][wave * 16 + l15][0];  // 40 bf16 = 5 v8bf
    v8bf alo = arow[half];        // k = half*8 .. +7
    v8bf ahi = arow[2 + half];    // k = 16+half*8 .. +7
    v16bf af = __builtin_shufflevector(alo, ahi,
        0,1,2,3,4,5,6,7,8,9,10,11,12,13,14,15);

    // ---- batch-load all 8 B fragments
    v16bf bfv[8];
#pragma unroll
    for (int tn = 0; tn < 8; ++tn) {
      const v8bf* brow = (const v8bf*)&Bt[cur][tn * 16 + l15][0];  // 56 bf16 = 7 v8bf
      v8bf blo = brow[half];
      v8bf bhi = brow[2 + half];
      bfv[tn] = __builtin_shufflevector(blo, bhi,
          0,1,2,3,4,5,6,7,8,9,10,11,12,13,14,15);
    }
    __builtin_amdgcn_sched_barrier(0);

    // ---- 8 back-to-back WMMAs (single dscnt wait ahead of the stream)
#pragma unroll
    for (int tn = 0; tn < 8; ++tn) {
      acc[tn] = __builtin_amdgcn_wmma_f32_16x16x32_bf16(
          false, af, false, bfv[tn], (short)0, acc[tn], false, false);
    }
    __builtin_amdgcn_sched_barrier(0);

    if (hasnext) lstore(cur ^ 1);        // s_wait_loadcnt happens here
    __syncthreads();
  }

  // ---- C/D layout: VGPR r -> M = r + half*8 ; N = lane&15
#pragma unroll
  for (int tn = 0; tn < 8; ++tn) {
#pragma unroll
    for (int r = 0; r < 8; ++r) {
      int m = m0 + wave * 16 + r + half * 8;
      int n = n0 + tn * 16 + l15;
      C[(size_t)m * N + n] = acc[tn][r];
    }
  }
}

// =====================================================================
// 2) beta / glog:  one block per (b,t); 12 warps -> 6 heads x {a,b}
// =====================================================================
__global__ __launch_bounds__(384)
void gdn_ab_kernel(const float* __restrict__ x, const float* __restrict__ w_a,
                   const float* __restrict__ w_b, const float* __restrict__ A_log,
                   const float* __restrict__ dt_bias,
                   float* __restrict__ beta, float* __restrict__ glog) {
  const int row  = blockIdx.x;          // b*T + t
  const int lane = threadIdx.x & 31;
  const int w    = threadIdx.x >> 5;    // 0..11
  const int h     = w >> 1;
  const int which = w & 1;              // 0 = a (glog), 1 = b (beta)
  const float* wm = which ? w_b : w_a;
  float acc = 0.0f;
  const float* xr = x + (size_t)row * HID_;
  for (int i = lane; i < HID_; i += 32) acc += xr[i] * wm[(size_t)i * NH_ + h];
  acc = xor_reduce32(acc, lane);
  if (lane == 0) {
    if (which) beta[(size_t)row * NH_ + h] = dev_sigmoid(acc);
    else       glog[(size_t)row * NH_ + h] = -expf(A_log[h]) * dev_softplus(acc + dt_bias[h]);
  }
}

// =====================================================================
// 3) depthwise causal conv (K=4) + SiLU
// =====================================================================
__global__ __launch_bounds__(256)
void gdn_conv_silu(const float* __restrict__ in, const float* __restrict__ w,
                   float* __restrict__ out, int C, int total) {
  int idx = blockIdx.x * 256 + threadIdx.x;
  if (idx >= total) return;
  int c  = idx % C;
  int bt = idx / C;
  int t  = bt % T_;
  float acc = 0.0f;
#pragma unroll
  for (int j = 0; j < CONV_; ++j) {
    int tt = t - (CONV_ - 1) + j;
    if (tt >= 0) acc += in[(size_t)(bt + (tt - t)) * C + c] * w[c * CONV_ + j];
  }
  out[idx] = dev_silu(acc);
}

// =====================================================================
// 4) l2norm over DK per (b,t,h), in place
// =====================================================================
__global__ __launch_bounds__(256)
void gdn_l2norm(float* __restrict__ p) {
  const int row = blockIdx.x;                 // (b*T+t)*NH + h
  const int bt = row / NH_, h = row % NH_;
  const size_t base = (size_t)bt * KEYD + (size_t)h * DK_;
  const int tid = threadIdx.x, lane = tid & 31, warp = tid >> 5;
  __shared__ float wsum[8];
  float v = p[base + tid];
  float ss = xor_reduce32(v * v, lane);
  if (lane == 0) wsum[warp] = ss;
  __syncthreads();
  float total = 0.0f;
#pragma unroll
  for (int i = 0; i < 8; ++i) total += wsum[i];
  float n = sqrtf(total);
  p[base + tid] = v / fmaxf(n, 1e-6f);
}

// =====================================================================
// 5) gated delta rule: 1 block per (b,h), state S[256][512] in registers.
//    thread (kk=tid&15, vv=tid>>4) owns S[kk*16..+16][vv*8..+8] (128 f32).
//    DK reductions via ds_bpermute xor inside 16-lane groups.
// =====================================================================
__global__ __launch_bounds__(1024)
void gdn_delta_rule(const float* __restrict__ q, const float* __restrict__ k,
                    const float* __restrict__ v, const float* __restrict__ glog,
                    const float* __restrict__ beta, float* __restrict__ o) {
  const int bh = blockIdx.x;            // 0..B*NH-1
  const int b = bh / NH_, h = bh % NH_;
  const int tid  = threadIdx.x;
  const int lane = tid & 31;
  const int kk = tid & 15;              // 16 k-groups (inside half-wave)
  const int vv = tid >> 4;              // 64 v-groups
  const int kbase = kk * 16;
  const int vbase = vv * 8;

  __shared__ float lq[DK_], lk[DK_], lv[DV_];
  __shared__ float lgb[2];

  float s[16][8];
#pragma unroll
  for (int i = 0; i < 16; ++i)
#pragma unroll
    for (int j = 0; j < 8; ++j) s[i][j] = 0.0f;

  for (int t = 0; t < T_; ++t) {
    const size_t rq = ((size_t)(b * T_ + t)) * KEYD + (size_t)h * DK_;
    const size_t rv = ((size_t)(b * T_ + t)) * VALD + (size_t)h * DV_;
    // cooperative stage: exactly one float per thread
    if (tid < DK_)            lq[tid]          = q[rq + tid];
    else if (tid < 2 * DK_)   lk[tid - DK_]    = k[rq + (tid - DK_)];
    else                      lv[tid - 2*DK_]  = v[rv + (tid - 2 * DK_)];
    if (tid == 0) {
      lgb[0] = expf(glog[(size_t)(b * T_ + t) * NH_ + h]);
      lgb[1] = beta[(size_t)(b * T_ + t) * NH_ + h];
    }
    __syncthreads();

    const float g  = lgb[0];
    const float bt = lgb[1];
    float kt[16], qt[16], vt[8];
#pragma unroll
    for (int i = 0; i < 16; ++i) { kt[i] = lk[kbase + i]; qt[i] = lq[kbase + i]; }
#pragma unroll
    for (int j = 0; j < 8; ++j)  vt[j] = lv[vbase + j];

    // decay + partial k^T (g*S)
    float p[8];
#pragma unroll
    for (int j = 0; j < 8; ++j) p[j] = 0.0f;
#pragma unroll
    for (int i = 0; i < 16; ++i)
#pragma unroll
      for (int j = 0; j < 8; ++j) { s[i][j] *= g; p[j] += kt[i] * s[i][j]; }
#pragma unroll
    for (int j = 0; j < 8; ++j) p[j] = xor_reduce16(p[j], lane);   // full kS on all 16 lanes

    float u[8];
#pragma unroll
    for (int j = 0; j < 8; ++j) u[j] = bt * (vt[j] - p[j]);

    // rank-1 update + partial q^T S_new
    float op[8];
#pragma unroll
    for (int j = 0; j < 8; ++j) op[j] = 0.0f;
#pragma unroll
    for (int i = 0; i < 16; ++i)
#pragma unroll
      for (int j = 0; j < 8; ++j) { s[i][j] += kt[i] * u[j]; op[j] += qt[i] * s[i][j]; }
#pragma unroll
    for (int j = 0; j < 8; ++j) op[j] = xor_reduce16(op[j], lane) * SCALE_;

    if (kk == 0) {
#pragma unroll
      for (int j = 0; j < 8; ++j) o[rv + vbase + j] = op[j];
    }
    __syncthreads();   // protect LDS stage buffers for next step
  }
}

// =====================================================================
// 6) gated RMSNorm: o = rmsnorm(o) * norm_w * silu(g_out)  (in place)
// =====================================================================
__global__ __launch_bounds__(256)
void gdn_rmsnorm_gate(float* __restrict__ o, const float* __restrict__ gout,
                      const float* __restrict__ norm_w) {
  const int row = blockIdx.x;               // (b*T+t)*NH + h
  const int bt = row / NH_, h = row % NH_;
  const size_t base = (size_t)bt * VALD + (size_t)h * DV_;
  const int tid = threadIdx.x, lane = tid & 31, warp = tid >> 5;
  __shared__ float wsum[8];
  float x0 = o[base + tid];
  float x1 = o[base + tid + 256];
  float ss = xor_reduce32(x0 * x0 + x1 * x1, lane);
  if (lane == 0) wsum[warp] = ss;
  __syncthreads();
  float total = 0.0f;
#pragma unroll
  for (int i = 0; i < 8; ++i) total += wsum[i];
  float r = rsqrtf(total / (float)DV_ + EPS_);
  o[base + tid]       = x0 * r * norm_w[tid]       * dev_silu(gout[base + tid]);
  o[base + tid + 256] = x1 * r * norm_w[tid + 256] * dev_silu(gout[base + tid + 256]);
}

// =====================================================================
// launcher
// =====================================================================
extern "C" void kernel_launch(void* const* d_in, const int* in_sizes, int n_in,
                              void* d_out, int out_size, void* d_ws, size_t ws_size,
                              hipStream_t stream) {
  (void)in_sizes; (void)n_in; (void)out_size; (void)ws_size;
  const float* x       = (const float*)d_in[0];
  const float* w_q     = (const float*)d_in[1];
  const float* w_k     = (const float*)d_in[2];
  const float* w_v     = (const float*)d_in[3];
  const float* w_a     = (const float*)d_in[4];
  const float* w_b     = (const float*)d_in[5];
  const float* w_g     = (const float*)d_in[6];
  const float* w_o     = (const float*)d_in[7];
  const float* conv_q  = (const float*)d_in[8];
  const float* conv_k  = (const float*)d_in[9];
  const float* conv_v  = (const float*)d_in[10];
  const float* A_log   = (const float*)d_in[11];
  const float* dt_bias = (const float*)d_in[12];
  const float* norm_w  = (const float*)d_in[13];
  float* out = (float*)d_out;

  float* ws = (float*)d_ws;
  size_t off = 0;
  float* qp   = ws + off; off += (size_t)BT_ * KEYD;   // q projection
  float* kp   = ws + off; off += (size_t)BT_ * KEYD;   // k projection
  float* vp   = ws + off; off += (size_t)BT_ * VALD;   // v projection
  float* qc   = ws + off; off += (size_t)BT_ * KEYD;   // conv q
  float* kc   = ws + off; off += (size_t)BT_ * KEYD;   // conv k
  float* vc   = ws + off; off += (size_t)BT_ * VALD;   // conv v
  float* beta = ws + off; off += (size_t)BT_ * NH_;
  float* glog = ws + off; off += (size_t)BT_ * NH_;
  // aliases (safe: strictly ordered on one stream):
  float* obuf = qp;   // qp+kp contiguous = BT*2*KEYD = BT*VALD floats; free after conv
  float* gout = vp;   // vp free after conv_v consumes it

  dim3 blk128(128);
  // --- projections (WMMA bf16 GEMMs)
  gdn_gemm_bf16_wmma<<<dim3(KEYD / GTN, BT_ / GTM), blk128, 0, stream>>>(x, w_q, qp, BT_, KEYD, HID_);
  gdn_gemm_bf16_wmma<<<dim3(KEYD / GTN, BT_ / GTM), blk128, 0, stream>>>(x, w_k, kp, BT_, KEYD, HID_);
  gdn_gemm_bf16_wmma<<<dim3(VALD / GTN, BT_ / GTM), blk128, 0, stream>>>(x, w_v, vp, BT_, VALD, HID_);
  // --- beta / glog
  gdn_ab_kernel<<<BT_, 384, 0, stream>>>(x, w_a, w_b, A_log, dt_bias, beta, glog);
  // --- short conv + SiLU
  {
    int tq = BT_ * KEYD, tv = BT_ * VALD;
    gdn_conv_silu<<<(tq + 255) / 256, 256, 0, stream>>>(qp, conv_q, qc, KEYD, tq);
    gdn_conv_silu<<<(tq + 255) / 256, 256, 0, stream>>>(kp, conv_k, kc, KEYD, tq);
    gdn_conv_silu<<<(tv + 255) / 256, 256, 0, stream>>>(vp, conv_v, vc, VALD, tv);
  }
  // --- gate projection (into vp region, now free)
  gdn_gemm_bf16_wmma<<<dim3(VALD / GTN, BT_ / GTM), blk128, 0, stream>>>(x, w_g, gout, BT_, VALD, HID_);
  // --- l2 norm q,k
  gdn_l2norm<<<BT_ * NH_, 256, 0, stream>>>(qc);
  gdn_l2norm<<<BT_ * NH_, 256, 0, stream>>>(kc);
  // --- sequential gated delta rule (register-resident state)
  gdn_delta_rule<<<B_ * NH_, 1024, 0, stream>>>(qc, kc, vc, glog, beta, obuf);
  // --- gated RMSNorm (in place on obuf)
  gdn_rmsnorm_gate<<<BT_ * NH_, 256, 0, stream>>>(obuf, gout, norm_w);
  // --- output projection
  gdn_gemm_bf16_wmma<<<dim3(HID_ / GTN, BT_ / GTM), blk128, 0, stream>>>(obuf, w_o, out, BT_, HID_, VALD);
}
